// StageAttention_88691074663057
// MI455X (gfx1250) — matrix operand order repaired and checked by reading, not compiled, not run on hardware
//
#include <hip/hip_runtime.h>
#include <hip/hip_bf16.h>

// ---------------------------------------------------------------------------
// 8x8 windowed MHA on MI455X (gfx1250), wave32 + v_wmma_f32_16x16x32_bf16.
// One block (256 thr = 8 waves) per window; wave w owns head w end-to-end.
// Roofline: 155 GFLOP / 0.54 GB HBM => AI ~287 FLOP/B: matrix pipe required.
// Round 3: (a) pad all LDS row strides so row stride % 32B == 16B -> A/B
// fragment ds_load_b128 across 16 rows tiles all 64 banks (was 16-way
// conflicted); (b) keep 12 accumulators live in QKV GEMM (3 N-tiles x 4
// M-tiles) and 8 in proj, cutting A-fragment LDS traffic 3x / 2x.
// ---------------------------------------------------------------------------

#define DIMC     256
#define NHEADS   8
#define HEADD    32
#define WIN      8
#define TOK      64          // tokens per window (8*8)
#define IMG      128
#define SCALE    0.17677669529663687f   // 32^-0.5

// Padded LDS row strides (elements): row stride in dwords ≡ 4 (mod 8)
#define TOKS     264         // tok / attn_out rows (bf16): 528B = 132 dw
#define QS       40          // q / k rows (bf16): 80B = 20 dw
#define VS       72          // vT rows (bf16): 144B = 36 dw
#define PS       72          // probs rows (bf16)
#define OS       68          // outf rows (f32): 272B = 68 dw

typedef __attribute__((ext_vector_type(16))) __bf16 v16bf;
typedef __attribute__((ext_vector_type(8)))  __bf16 v8bf;
typedef __attribute__((ext_vector_type(8)))  float  v8f;

__device__ __forceinline__ v8f wmma_bf16(v16bf a, v16bf b, v8f c) {
    return __builtin_amdgcn_wmma_f32_16x16x32_bf16(
        /*neg_a=*/false, a, /*neg_b=*/false, b,
        /*c_mod=*/(short)0, c, /*reuse_a=*/false, /*reuse_b=*/false);
}

__device__ __forceinline__ v8f splat8(float x) {
    v8f r;
#pragma unroll
    for (int i = 0; i < 8; ++i) r[i] = x;
    return r;
}

// A-fragment (16x32 bf16, MxK): lane m = M row; K octets per ISA layout:
//   elems 0..7  <-> K = 8*hl + 0..7,  elems 8..15 <-> K = 16 + 8*hl + 0..7
__device__ __forceinline__ v16bf load_afrag(const __bf16* row, int kc, int hl) {
    const v8bf lo = *(const v8bf*)(row + kc + 8 * hl);
    const v8bf hi = *(const v8bf*)(row + kc + 16 + 8 * hl);
    return __builtin_shufflevector(lo, hi, 0,1,2,3,4,5,6,7,8,9,10,11,12,13,14,15);
}

// B-fragment (32x16 bf16, KxN): lane n = N col; elems 0..7 <-> K = 8*hl + 0..7,
// elems 8..15 <-> K = 16 + 8*hl + 0..7 (matches A). Built from two 16B loads so only
// 16B alignment is assumed (padded strides are 16B multiples).
__device__ __forceinline__ v16bf load_bfrag(const __bf16* row, int kc, int hl) {
    const v8bf lo = *(const v8bf*)(row + kc + 8 * hl);
    const v8bf hi = *(const v8bf*)(row + kc + 16 + 8 * hl);
    return __builtin_shufflevector(lo, hi, 0,1,2,3,4,5,6,7,8,9,10,11,12,13,14,15);
}

// ---------------------------------------------------------------------------
// Pre-kernel: convert fp32 weights to bf16 into workspace (every launch;
// deterministic; 0.5 MB stays hot in the 192 MB L2 across all 4096 blocks).
// ---------------------------------------------------------------------------
__global__ void wconv_kernel(const float* __restrict__ wq,
                             const float* __restrict__ wp,
                             __bf16* __restrict__ oq,
                             __bf16* __restrict__ op) {
    int i = blockIdx.x * blockDim.x + threadIdx.x;
    if (i < 3 * DIMC * DIMC) oq[i] = (__bf16)wq[i];
    if (i < DIMC * DIMC)     op[i] = (__bf16)wp[i];
}

// ---------------------------------------------------------------------------
// LDS map (221 KB dynamic):
//   [0      ,  33792) tok  bf16[64][264]  -> reused as attn_out
//   [33792  ,  74752) q    bf16[8][64][40]
//   [74752  , 115712) k    bf16[8][64][40]
//   [115712 , 152576) vT   bf16[8][32][72]
//   [152576 , 226304) probs bf16[8][64][72]
//   [33792  , 103424) outf f32[256][68]   (aliases q/k, live only in proj)
// ---------------------------------------------------------------------------
__global__ __launch_bounds__(256) void win_attn_kernel(
        const float* __restrict__ x,
        const float* __restrict__ b_qkv,
        const float* __restrict__ b_proj,
        const __bf16* __restrict__ wq,    // 768x256 bf16 (row-major, = w_qkv)
        const __bf16* __restrict__ wp,    // 256x256 bf16 (row-major, = w_proj)
        float* __restrict__ out) {
    extern __shared__ __align__(32) unsigned char smem[];
    __bf16* tok   = (__bf16*)(smem);             // [64][TOKS]
    __bf16* qb    = (__bf16*)(smem + 33792);     // [8][64][QS]
    __bf16* kb    = (__bf16*)(smem + 74752);     // [8][64][QS]
    __bf16* vT    = (__bf16*)(smem + 115712);    // [8][32][VS]
    __bf16* pb    = (__bf16*)(smem + 152576);    // [8][64][PS]
    float*  outf  = (float*)(smem + 33792);      // [256][OS]
    __bf16* aout  = tok;                         // attn output reuses tok

    const int tid = threadIdx.x;
    const int wv  = tid >> 5;        // wave id == head id
    const int ln  = tid & 31;
    const int lnl = ln & 15;         // N / column lane
    const int hl  = ln >> 4;         // lane half

    const int blk = blockIdx.x;      // 16 * 16 * 16 windows
    const int b   = blk >> 8;
    const int hn  = (blk >> 4) & 15;
    const int wn  = blk & 15;
    const int h0  = hn * WIN, w0 = wn * WIN;

    // ---- Stage A: gather window, convert to bf16 tokens[t][c] --------------
    for (int t = tid; t < DIMC * WIN; t += 256) {          // (c, row) tasks
        const int c = t >> 3, r = t & 7;
        const float* src = x + ((long)(b * DIMC + c) << 14) + (h0 + r) * IMG + w0;
        const float4 a0 = *(const float4*)(src);
        const float4 a1 = *(const float4*)(src + 4);
        const int tb = r * WIN;
        tok[(tb + 0) * TOKS + c] = (__bf16)a0.x;
        tok[(tb + 1) * TOKS + c] = (__bf16)a0.y;
        tok[(tb + 2) * TOKS + c] = (__bf16)a0.z;
        tok[(tb + 3) * TOKS + c] = (__bf16)a0.w;
        tok[(tb + 4) * TOKS + c] = (__bf16)a1.x;
        tok[(tb + 5) * TOKS + c] = (__bf16)a1.y;
        tok[(tb + 6) * TOKS + c] = (__bf16)a1.z;
        tok[(tb + 7) * TOKS + c] = (__bf16)a1.w;
    }
    __syncthreads();

    // ---- Stage B: QKV GEMM; wave w computes q/k/v columns of head w --------
    // out[t][o] = sum_c tok[t][c]*w_qkv[o][c] + b_qkv[o]
    // 2 groups of 3 N-tiles: 12 live accumulators -> 4 A + 3 B frags / K-step.
    for (int g = 0; g < 2; ++g) {
        const __bf16* brow[3];
        v8f acc[3][4];
#pragma unroll
        for (int j = 0; j < 3; ++j) {
            const int nt  = g * 3 + j;
            const int oc  = (nt >> 1) * DIMC + wv * HEADD + (nt & 1) * 16 + lnl;
            const float bias = b_qkv[oc];
            brow[j] = wq + (long)oc * DIMC;
#pragma unroll
            for (int mt = 0; mt < 4; ++mt) acc[j][mt] = splat8(bias);
        }
#pragma unroll
        for (int kc = 0; kc < DIMC; kc += 32) {
            v16bf bf[3];
#pragma unroll
            for (int j = 0; j < 3; ++j) bf[j] = load_bfrag(brow[j], kc, hl);
            v16bf af[4];
#pragma unroll
            for (int mt = 0; mt < 4; ++mt)
                af[mt] = load_afrag(tok + (mt * 16 + lnl) * TOKS, kc, hl);
#pragma unroll
            for (int j = 0; j < 3; ++j)
#pragma unroll
                for (int mt = 0; mt < 4; ++mt)
                    acc[j][mt] = wmma_bf16(af[mt], bf[j], acc[j][mt]);
        }
#pragma unroll
        for (int j = 0; j < 3; ++j) {
            const int nt  = g * 3 + j;
            const int sel = nt >> 1, nn = nt & 1;
            const int col = nn * 16 + lnl;       // 0..31 within head
#pragma unroll
            for (int mt = 0; mt < 4; ++mt)
#pragma unroll
                for (int r = 0; r < 8; ++r) {
                    const int m = mt * 16 + r + 8 * hl;
                    const __bf16 v = (__bf16)acc[j][mt][r];
                    if (sel == 0)      qb[(wv * TOK + m) * QS + col] = v;
                    else if (sel == 1) kb[(wv * TOK + m) * QS + col] = v;
                    else               vT[(wv * HEADD + col) * VS + m] = v;
                }
        }
    }
    __syncthreads();

    // ---- Stage C: scores + softmax for head wv (all intra-wave) ------------
    const __bf16* qh = qb + wv * TOK * QS;
    const __bf16* kh = kb + wv * TOK * QS;
    const __bf16* vh = vT + wv * HEADD * VS;
    __bf16*       ph = pb + wv * TOK * PS;
    for (int mt = 0; mt < 4; ++mt) {
        const v16bf aq = load_afrag(qh + (mt * 16 + lnl) * QS, 0, hl);
        v16bf bk[4];
#pragma unroll
        for (int nt = 0; nt < 4; ++nt)
            bk[nt] = load_bfrag(kh + (nt * 16 + lnl) * QS, 0, hl);
        v8f s[4];
#pragma unroll
        for (int nt = 0; nt < 4; ++nt)
            s[nt] = wmma_bf16(aq, bk[nt], splat8(0.0f));
#pragma unroll
        for (int r = 0; r < 8; ++r) {
            float mx = fmaxf(fmaxf(s[0][r], s[1][r]), fmaxf(s[2][r], s[3][r]));
#pragma unroll
            for (int off = 8; off; off >>= 1)
                mx = fmaxf(mx, __shfl_xor(mx, off, 32));   // 16-lane butterfly
            float sum = 0.f;
#pragma unroll
            for (int nt = 0; nt < 4; ++nt) {
                const float e = __expf((s[nt][r] - mx) * SCALE);
                s[nt][r] = e;  sum += e;
            }
#pragma unroll
            for (int off = 8; off; off >>= 1)
                sum += __shfl_xor(sum, off, 32);
            const float inv = 1.0f / sum;
            const int m = mt * 16 + r + 8 * hl;
#pragma unroll
            for (int nt = 0; nt < 4; ++nt)
                ph[m * PS + nt * 16 + lnl] = (__bf16)(s[nt][r] * inv);
        }
    }

    // ---- Stage C2: attn @ V for head wv ------------------------------------
    for (int mt = 0; mt < 4; ++mt) {
        v16bf ap[2];
#pragma unroll
        for (int ks = 0; ks < 2; ++ks)
            ap[ks] = load_afrag(ph + (mt * 16 + lnl) * PS, ks * 32, hl);
#pragma unroll
        for (int nc = 0; nc < 2; ++nc) {
            v16bf bv[2];
#pragma unroll
            for (int ks = 0; ks < 2; ++ks)
                bv[ks] = load_bfrag(vh + (nc * 16 + lnl) * VS, ks * 32, hl);
            v8f acc = splat8(0.0f);
#pragma unroll
            for (int ks = 0; ks < 2; ++ks)
                acc = wmma_bf16(ap[ks], bv[ks], acc);
#pragma unroll
            for (int r = 0; r < 8; ++r) {
                const int m = mt * 16 + r + 8 * hl;
                aout[m * TOKS + wv * HEADD + nc * 16 + lnl] = (__bf16)acc[r];
            }
        }
    }
    __syncthreads();

    // ---- Stage D: projection; wave w owns output cols [32w, 32w+32) --------
    // All 8 accumulators live -> 4 A + 2 B frags per K-step.
    {
        const int oc0 = wv * 2 * 16 + lnl;       // nt2 = 0 column
        const int oc1 = oc0 + 16;                // nt2 = 1 column
        const __bf16* brow0 = wp + (long)oc0 * DIMC;
        const __bf16* brow1 = wp + (long)oc1 * DIMC;
        v8f acc[2][4];
#pragma unroll
        for (int mt = 0; mt < 4; ++mt) {
            acc[0][mt] = splat8(b_proj[oc0]);
            acc[1][mt] = splat8(b_proj[oc1]);
        }
#pragma unroll
        for (int kc = 0; kc < DIMC; kc += 32) {
            const v16bf bf0 = load_bfrag(brow0, kc, hl);
            const v16bf bf1 = load_bfrag(brow1, kc, hl);
            v16bf af[4];
#pragma unroll
            for (int mt = 0; mt < 4; ++mt)
                af[mt] = load_afrag(aout + (mt * 16 + lnl) * TOKS, kc, hl);
#pragma unroll
            for (int mt = 0; mt < 4; ++mt) {
                acc[0][mt] = wmma_bf16(af[mt], bf0, acc[0][mt]);
                acc[1][mt] = wmma_bf16(af[mt], bf1, acc[1][mt]);
            }
        }
#pragma unroll
        for (int j = 0; j < 2; ++j) {
            const int oc = j ? oc1 : oc0;
#pragma unroll
            for (int mt = 0; mt < 4; ++mt)
#pragma unroll
                for (int r = 0; r < 8; ++r) {
                    const int m = mt * 16 + r + 8 * hl;
                    outf[oc * OS + m] = acc[j][mt][r];   // [chan][token] f32
                }
        }
    }
    __syncthreads();

    // ---- Stage E: coalesced-ish scatter back to BCHW (32B row chunks) ------
    for (int t = tid; t < DIMC * WIN; t += 256) {
        const int c = t >> 3, r = t & 7;
        const float4 v0 = *(const float4*)&outf[c * OS + r * WIN];
        const float4 v1 = *(const float4*)&outf[c * OS + r * WIN + 4];
        float* dst = out + ((long)(b * DIMC + c) << 14) + (h0 + r) * IMG + w0;
        *(float4*)(dst)     = v0;
        *(float4*)(dst + 4) = v1;
    }
}

extern "C" void kernel_launch(void* const* d_in, const int* in_sizes, int n_in,
                              void* d_out, int out_size, void* d_ws, size_t ws_size,
                              hipStream_t stream) {
    const float* x      = (const float*)d_in[0];
    const float* w_qkv  = (const float*)d_in[1];
    const float* b_qkv  = (const float*)d_in[2];
    const float* w_proj = (const float*)d_in[3];
    const float* b_proj = (const float*)d_in[4];
    float* out = (float*)d_out;

    __bf16* wq_bf = (__bf16*)d_ws;                           // 768*256 bf16
    __bf16* wp_bf = (__bf16*)((char*)d_ws + 3 * DIMC * DIMC * sizeof(__bf16));

    wconv_kernel<<<(3 * DIMC * DIMC + 255) / 256, 256, 0, stream>>>(
        w_qkv, w_proj, wq_bf, wp_bf);

    const int smem_bytes = 226304;   // 221 KB dynamic LDS (<= 320 KB/WGP)
    hipFuncSetAttribute((const void*)win_attn_kernel,
                        hipFuncAttributeMaxDynamicSharedMemorySize, smem_bytes);

    win_attn_kernel<<<4096, 256, smem_bytes, stream>>>(
        x, b_qkv, b_proj, wq_bf, wp_bf, out);
}
